// EdgeBlockConcat_84104049590405
// MI455X (gfx1250) — matrix operand order repaired and checked by reading, not compiled, not run on hardware
//
#include <hip/hip_runtime.h>
#include <hip/hip_bf16.h>

typedef __attribute__((ext_vector_type(16))) _Float16 v16h;
typedef __attribute__((ext_vector_type(8)))  _Float16 v8h;
typedef __attribute__((ext_vector_type(8)))  float    v8f;

#define E_EDGES     500000
#define NN_NODES    50000
#define DDIM        128
#define TILES       (E_EDGES / 16)          // 31250 (exact)
#define FRAG_HALVES 512                      // 32 lanes * 16 halves
#define W1_FRAGS    (12 * 8)                 // K-tiles x N-tiles
#define W2_FRAGS    (4 * 8)
#define W1_HALVES   (W1_FRAGS * FRAG_HALVES) // 49152
#define W2_HALVES   (W2_FRAGS * FRAG_HALVES) // 16384
#define WT_HALVES   (W1_HALVES + W2_HALVES)  // 65536 halves = 128 KB

// ---------------------------------------------------------------------------
// Prologue: convert W1/W2 (f32, row-major KxN) into f16 B-fragments laid out
// exactly as V_WMMA_F32_16X16X32_F16 expects its B operand per lane:
//   lane l (0..15): N = l,      K = kt*32 + j (j<8), kt*32 + 8 + j (j>=8)
//   lane l (16..31): N = l-16,  K = kt*32 + 8 + j (j<8), kt*32 + 16 + j (j>=8)
// Fragment (kt,nt) stored contiguously: ws[frag*512 + lane*16 + j].
// ---------------------------------------------------------------------------
__global__ void prep_weights_kernel(const float* __restrict__ W1,
                                    const float* __restrict__ W2,
                                    _Float16* __restrict__ ws) {
    int gid = blockIdx.x * blockDim.x + threadIdx.x;
    if (gid >= WT_HALVES) return;
    const float* W;
    int g;
    if (gid < W1_HALVES) { W = W1; g = gid; }
    else                 { W = W2; g = gid - W1_HALVES; }
    int frag = g >> 9;
    int l    = (g >> 4) & 31;
    int j    = g & 15;
    int kt   = frag >> 3;
    int nt   = frag & 7;
    int k    = kt * 32 + ((l < 16) ? 0 : 8) + j + ((j < 8) ? 0 : 8);
    int n    = nt * 16 + (l & 15);
    ws[gid]  = (_Float16)W[k * DDIM + n];
}

// ---------------------------------------------------------------------------
// Main kernel: one wave per 16-edge tile (grid-stride). Weights staged to LDS.
// ---------------------------------------------------------------------------
__global__ void __launch_bounds__(256) edge_mlp_kernel(
    const float* __restrict__ efeat, const float* __restrict__ nfeat,
    const int*   __restrict__ src,   const int*   __restrict__ dst,
    const float* __restrict__ b1,    const float* __restrict__ b2,
    const float* __restrict__ gamma, const float* __restrict__ beta,
    const _Float16* __restrict__ wsW, float* __restrict__ out, int numTiles)
{
    extern __shared__ _Float16 smem[];
    _Float16* ldsW = smem;                         // 65536 halves (W1 then W2)
    _Float16* ldsH = smem + WT_HALVES;             // 8 waves * 16*128 halves

    const int tid  = threadIdx.x;
    const int lane = tid & 31;
    const int wave = tid >> 5;

    // Cooperative stage of pre-swizzled weight fragments into LDS (128 KB).
    {
        const uint4* s = (const uint4*)wsW;
        uint4*       d = (uint4*)ldsW;
        for (int i = tid; i < (WT_HALVES / 8); i += 256) d[i] = s[i];
    }
    __syncthreads();

    const int m  = lane & 15;       // row (GEMM A) / column (B,C,D) within tile
    const int hi = lane >> 4;       // lane half
    const int o8 = hi ? 8 : 0;      // A-fragment K base offset for this half

    // Per-lane broadcast vectors for bias / LN affine (index n = nt*16 + m)
    float b1v[8], b2v[8], gv[8], bv[8];
#pragma unroll
    for (int nt = 0; nt < 8; ++nt) {
        int n   = nt * 16 + m;
        b1v[nt] = b1[n];
        b2v[nt] = b2[n];
        gv[nt]  = gamma[n];
        bv[nt]  = beta[n];
    }

    _Float16* hbuf = ldsH + wave * (16 * 128);     // private per wave

    const int waveGlobal = blockIdx.x * 8 + wave;
    const int waveStride = gridDim.x * 8;

    for (int t = waveGlobal; t < numTiles; t += waveStride) {
        const int eBase = t * 16;
        const int eRow  = eBase + m;
        const int sIdx  = src[eRow];
        const int dIdx  = dst[eRow];
        const float* r0 = efeat + eRow * DDIM;     // cat dims   0..127
        const float* r1 = nfeat + sIdx * DDIM;     // cat dims 128..255
        const float* r2 = nfeat + dIdx * DDIM;     // cat dims 256..383

        // ---------------- GEMM1: [16,384] x [384,128] -> h [16,128] --------
        v8f acc[8] = {};
#pragma unroll
        for (int kt = 0; kt < 12; ++kt) {
            const float* rp  = (kt < 4) ? r0 : (kt < 8) ? r1 : r2;
            const int    loc = (kt & 3) * 32;
            // ISA 16-bit A 16x32 layout: this lane needs K = loc+o8..+7 and
            // K = loc+o8+16..+23 (coalesced float4 pairs).
            const float4 x0 = *(const float4*)(rp + loc + o8);
            const float4 x1 = *(const float4*)(rp + loc + o8 + 4);
            const float4 x2 = *(const float4*)(rp + loc + o8 + 16);
            const float4 x3 = *(const float4*)(rp + loc + o8 + 20);
            v16h a;
            a[0]=(_Float16)x0.x; a[1]=(_Float16)x0.y; a[2]=(_Float16)x0.z; a[3]=(_Float16)x0.w;
            a[4]=(_Float16)x1.x; a[5]=(_Float16)x1.y; a[6]=(_Float16)x1.z; a[7]=(_Float16)x1.w;
            a[8]=(_Float16)x2.x; a[9]=(_Float16)x2.y; a[10]=(_Float16)x2.z; a[11]=(_Float16)x2.w;
            a[12]=(_Float16)x3.x; a[13]=(_Float16)x3.y; a[14]=(_Float16)x3.z; a[15]=(_Float16)x3.w;
#pragma unroll
            for (int nt = 0; nt < 8; ++nt) {
                const v16h b = *(const v16h*)(ldsW + (kt * 8 + nt) * FRAG_HALVES + lane * 16);
                acc[nt] = __builtin_amdgcn_wmma_f32_16x16x32_f16(
                    false, a, false, b, (short)0, acc[nt], false, false);
            }
        }

        // ------------- bias + SiLU, spill h to LDS in f16 (A-major) --------
#pragma unroll
        for (int nt = 0; nt < 8; ++nt) {
#pragma unroll
            for (int r = 0; r < 8; ++r) {
                float x = acc[nt][r] + b1v[nt];
                float s = x / (1.0f + __expf(-x));
                // D layout: row M = r + o8, col n = nt*16 + m
                hbuf[(r + o8) * 128 + nt * 16 + m] = (_Float16)s;
            }
        }
        asm volatile("s_wait_dscnt 0" ::: "memory");  // wave-private LDS: no barrier needed

        // ---------------- GEMM2: h [16,128] x W2 [128,128] -----------------
        v8f acc2[8] = {};
#pragma unroll
        for (int kt = 0; kt < 4; ++kt) {
            const v8h lo = *(const v8h*)(hbuf + m * 128 + kt * 32 + o8);
            const v8h hh = *(const v8h*)(hbuf + m * 128 + kt * 32 + o8 + 16);
            v16h a;
#pragma unroll
            for (int j = 0; j < 8; ++j) { a[j] = lo[j]; a[j + 8] = hh[j]; }
#pragma unroll
            for (int nt = 0; nt < 8; ++nt) {
                const v16h b = *(const v16h*)(ldsW + W1_HALVES + (kt * 8 + nt) * FRAG_HALVES + lane * 16);
                acc2[nt] = __builtin_amdgcn_wmma_f32_16x16x32_f16(
                    false, a, false, b, (short)0, acc2[nt], false, false);
            }
        }

        // -------- bias + LayerNorm + residual + store (per output row) ----
#pragma unroll
        for (int r = 0; r < 8; ++r) {
            float y[8];
            float sum = 0.0f;
#pragma unroll
            for (int nt = 0; nt < 8; ++nt) { y[nt] = acc2[nt][r] + b2v[nt]; sum += y[nt]; }
            // row M = r + o8 lives on one 16-lane half; masks 1/2/4/8 stay inside it
            sum += __shfl_xor(sum, 1);  sum += __shfl_xor(sum, 2);
            sum += __shfl_xor(sum, 4);  sum += __shfl_xor(sum, 8);
            const float mu = sum * (1.0f / 128.0f);
            float sq = 0.0f;
#pragma unroll
            for (int nt = 0; nt < 8; ++nt) { float dd = y[nt] - mu; sq += dd * dd; }
            sq += __shfl_xor(sq, 1);  sq += __shfl_xor(sq, 2);
            sq += __shfl_xor(sq, 4);  sq += __shfl_xor(sq, 8);
            const float rstd = rsqrtf(sq * (1.0f / 128.0f) + 1e-5f);

            const int eo = (eBase + r + o8) * DDIM;
#pragma unroll
            for (int nt = 0; nt < 8; ++nt) {
                const int n = nt * 16 + m;
                out[eo + n] = (y[nt] - mu) * rstd * gv[nt] + bv[nt] + efeat[eo + n];
            }
        }
    }
}

// ---------------------------------------------------------------------------
extern "C" void kernel_launch(void* const* d_in, const int* in_sizes, int n_in,
                              void* d_out, int out_size, void* d_ws, size_t ws_size,
                              hipStream_t stream) {
    (void)in_sizes; (void)n_in; (void)out_size; (void)ws_size;
    const float* efeat = (const float*)d_in[0];
    const float* nfeat = (const float*)d_in[1];
    const int*   src   = (const int*)d_in[2];
    const int*   dst   = (const int*)d_in[3];
    const float* W1    = (const float*)d_in[4];
    const float* b1    = (const float*)d_in[5];
    const float* W2    = (const float*)d_in[6];
    const float* b2    = (const float*)d_in[7];
    const float* gamma = (const float*)d_in[8];
    const float* beta  = (const float*)d_in[9];
    float*       out   = (float*)d_out;
    _Float16*    ws    = (_Float16*)d_ws;

    // Output tuple: (efeat_new [E,128], nfeat [N,128]) — passthrough copy.
    hipMemcpyAsync(out + (size_t)E_EDGES * DDIM, nfeat,
                   (size_t)NN_NODES * DDIM * sizeof(float),
                   hipMemcpyDeviceToDevice, stream);

    // 1) swizzle weights into WMMA B-fragment layout (f16) in workspace
    prep_weights_kernel<<<(WT_HALVES + 255) / 256, 256, 0, stream>>>(W1, W2, ws);

    // 2) main fused kernel
    const int    blocks = 512;                                   // grid-stride
    const size_t shmem  = (size_t)(WT_HALVES + 8 * 16 * 128) * sizeof(_Float16); // 160 KB
    edge_mlp_kernel<<<blocks, 256, shmem, stream>>>(
        efeat, nfeat, src, dst, b1, b2, gamma, beta, ws, out, TILES);
}
